// LSTMClassifier_87677462381300
// MI455X (gfx1250) — compile-verified, matrix operand order checked by reading, not run on hardware
//
#include <hip/hip_runtime.h>

typedef __attribute__((ext_vector_type(16))) _Float16 v16h;
typedef __attribute__((ext_vector_type(8)))  _Float16 v8h;
typedef __attribute__((ext_vector_type(8)))  float    v8f;

#define VOCAB 30000
#define EMB 256
#define HID 256
#define LAYERS 5
#define NLAB 10
#define BATCH 128
#define TLEN 512
#define BT (BATCH * TLEN)
#define G4H (4 * HID)
#define WPAD 40                       // LDS row stride in halves: 80B, 16B aligned, conflict-free b128
#define SBUF_BYTES (G4H * WPAD * 2)   // one staging buffer: 81920 B

// ---------- helpers ----------

__device__ __forceinline__ float sigm(float x) {
    return 1.0f / (1.0f + __expf(-x));
}
__device__ __forceinline__ float tanh_fast(float x) {
    float e = __expf(2.0f * x);
    return (e - 1.0f) / (e + 1.0f);
}

// async global->LDS 128-bit copy (CDNA5, ASYNCcnt-tracked)
__device__ __forceinline__ void async_g2l_b128(const _Float16* g, unsigned ldsoff) {
    asm volatile("global_load_async_to_lds_b128 %0, %1, off" :: "v"(ldsoff), "v"(g) : "memory");
}
#define ASYNC_WAIT_16() asm volatile("s_wait_asynccnt 16" ::: "memory")
#define ASYNC_WAIT_0()  asm volatile("s_wait_asynccnt 0"  ::: "memory")

// software grid barrier (all 64 blocks resident; phase/count zeroed per launch)
__device__ __forceinline__ void grid_sync(unsigned* count, unsigned* phase, unsigned* lp) {
    __syncthreads();
    if (threadIdx.x == 0) {
        unsigned target = ++(*lp);
        __threadfence();
        unsigned prev = atomicAdd(count, 1u);
        if (prev == gridDim.x - 1) {
            atomicExch(count, 0u);
            __threadfence();
            atomicAdd(phase, 1u);
        } else {
            while (atomicAdd(phase, 0u) < target) {
                __builtin_amdgcn_s_sleep(8);
            }
        }
    }
    __syncthreads();
}

// ---------- kernels ----------

// gather embedding rows, f32 -> f16. one thread = 8 elements.
__global__ void k_embed(const int* __restrict__ ids, const float* __restrict__ emb,
                        _Float16* __restrict__ X) {
    size_t i = (size_t)blockIdx.x * blockDim.x + threadIdx.x; // BT*EMB/8 threads
    size_t bt = i >> 5;                                       // 32 chunks of 8 per row
    int eo = (int)(i & 31) << 3;
    int id = ids[bt];
    const float4* s = (const float4*)(emb + (size_t)id * EMB + eo);
    float4 a = s[0], b = s[1];
    v8h o;
    o[0] = (_Float16)a.x; o[1] = (_Float16)a.y; o[2] = (_Float16)a.z; o[3] = (_Float16)a.w;
    o[4] = (_Float16)b.x; o[5] = (_Float16)b.y; o[6] = (_Float16)b.z; o[7] = (_Float16)b.w;
    *(v8h*)(X + bt * EMB + eo) = o;
}

// convert weights to f16; fuse biases
__global__ void k_cvtw(const float* __restrict__ wih, const float* __restrict__ whh,
                       const float* __restrict__ bih, const float* __restrict__ bhh,
                       _Float16* __restrict__ wihH, _Float16* __restrict__ whhH,
                       float* __restrict__ bsum) {
    size_t i = (size_t)blockIdx.x * blockDim.x + threadIdx.x; // LAYERS*G4H*HID threads
    if (i < (size_t)LAYERS * G4H * HID) {
        wihH[i] = (_Float16)wih[i];
        whhH[i] = (_Float16)whh[i];
    }
    if (i < (size_t)LAYERS * G4H) {
        bsum[i] = bih[i] + bhh[i];
    }
}

// P[t, b, n] = sum_k X[b*T+t, k] * W[n, k] + bsum[n], f16 out, [T][B][4H] layout.
// grid: BT/16 blocks of 256 threads (8 waves); wave w handles n-tiles [w*8, w*8+8).
// W K-chunks staged in LDS via async copies, double-buffered; B fragments SW-pipelined.
__global__ void __launch_bounds__(256, 1)
k_xproj(const _Float16* __restrict__ X, const _Float16* __restrict__ W,
        const float* __restrict__ bsum, _Float16* __restrict__ P) {
    __shared__ __align__(16) _Float16 sW[2][G4H * WPAD]; // 2 x 80KB

    const int tid   = threadIdx.x;
    const int lane  = tid & 31;
    const int wave  = tid >> 5;
    const int mt    = blockIdx.x;
    const int l16   = lane & 15;
    const int khalf = lane >> 4;
    const int mrow  = mt * 16 + l16;     // A-fragment row (bt index) for this lane
    const int ncol  = wave * 128 + l16;  // base N column (add ntl*16)

    const unsigned ldsBase = (unsigned)(uintptr_t)(const _Float16*)&sW[0][0];
    const char*    sGen    = (const char*)&sW[0][0]; // generic pointer for ds reads

    // async fill: thread covers (row = tid>>2 + 64*i, part = tid&3), i = 0..15
    const int frow = tid >> 2, fpart = tid & 3;
    const _Float16* gBase = W + (size_t)frow * HID + fpart * 8;
    const unsigned  lBase = ldsBase + (unsigned)((frow * WPAD + fpart * 8) * 2);
    auto fill = [&](int c, int buf) {
        const _Float16* g = gBase + c * 32;
        unsigned l = lBase + (unsigned)buf * SBUF_BYTES;
#pragma unroll
        for (int i = 0; i < 16; ++i) {
            async_g2l_b128(g, l);
            g += 64 * HID;            // 64 rows ahead
            l += 64 * WPAD * 2;
        }
    };

    fill(0, 0);

    // loop-invariant per-lane LDS byte offsets of the 8 B fragments (within one buffer)
    unsigned bbase[8];
#pragma unroll
    for (int ntl = 0; ntl < 8; ++ntl)
        bbase[ntl] = (unsigned)((((ncol + ntl * 16) * WPAD) + khalf * 8) * 2);

    v8f acc[8];
#pragma unroll
    for (int ntl = 0; ntl < 8; ++ntl) {
        float bv = bsum[ncol + ntl * 16];
#pragma unroll
        for (int v = 0; v < 8; ++v) acc[ntl][v] = bv;
    }

    union F { v16h h16; uint4 q[2]; };

    for (int c = 0; c < 8; ++c) {
        const int kk = c * 32;
        if (c < 7) { fill(c + 1, (c + 1) & 1); ASYNC_WAIT_16(); }
        else       { ASYNC_WAIT_0(); }
        __syncthreads();

        F A;
        const _Float16* xrow = X + (size_t)mrow * HID + kk + khalf * 8;
        A.q[0] = *(const uint4*)(xrow);
        A.q[1] = *(const uint4*)(xrow + 16);

        const char* sb = sGen + (unsigned)(c & 1) * SBUF_BYTES;

        // software pipeline: load fragment ntl+1 while WMMA consumes fragment ntl
        F Bf[2];
        Bf[0].q[0] = *(const uint4*)(sb + bbase[0]);
        Bf[0].q[1] = *(const uint4*)(sb + bbase[0] + 32);
#pragma unroll
        for (int ntl = 0; ntl < 8; ++ntl) {
            if (ntl < 7) {
                Bf[(ntl + 1) & 1].q[0] = *(const uint4*)(sb + bbase[ntl + 1]);
                Bf[(ntl + 1) & 1].q[1] = *(const uint4*)(sb + bbase[ntl + 1] + 32);
            }
            acc[ntl] = __builtin_amdgcn_wmma_f32_16x16x32_f16(
                false, A.h16, false, Bf[ntl & 1].h16, (short)0, acc[ntl], false, false);
        }
        __syncthreads(); // protect buffers before next fill overwrites
    }

    const int rowbase = mt * 16 + khalf * 8; // C/D layout: row = v + (lane>>4)*8
#pragma unroll
    for (int ntl = 0; ntl < 8; ++ntl) {
        int col = ncol + ntl * 16;
#pragma unroll
        for (int v = 0; v < 8; ++v) {
            int bt = rowbase + v;
            int b = bt >> 9, t = bt & (TLEN - 1);
            P[((size_t)t * BATCH + b) * G4H + col] = (_Float16)acc[ntl][v];
        }
    }
}

// persistent recurrent scan. 64 blocks x 256 threads = 512 waves = one 16x16 tile each.
// W_hh fragments are timestep-invariant -> hoisted into 64 VGPRs before the scan.
__global__ void __launch_bounds__(256, 1)
k_scan(const _Float16* __restrict__ P, const _Float16* __restrict__ Whh,
       _Float16* __restrict__ Hbuf, float* __restrict__ Cbuf,
       float* __restrict__ Gates, _Float16* __restrict__ Xout,
       unsigned* __restrict__ bar) {
    const int lane  = threadIdx.x & 31;
    const int wave  = threadIdx.x >> 5;
    const int gw    = blockIdx.x * 8 + wave; // 0..511
    const int mt    = gw >> 6;               // 0..7   (batch tile)
    const int nt    = gw & 63;               // 0..63  (gate-column tile)
    const int l16   = lane & 15;
    const int khalf = lane >> 4;
    const int mrow  = mt * 16 + l16;         // A row (batch index)
    const int ncol  = nt * 16 + l16;         // N column (gate index)
    const int crow0 = mt * 16 + khalf * 8;   // C/D row base

    unsigned* bcount = bar;
    unsigned* bphase = bar + 1;
    unsigned  lphase = 0;

    const int gtid = blockIdx.x * 256 + threadIdx.x; // 0..16383

    union F { v16h h16; uint4 q[2]; };

    // hoist all B fragments (W_hh rows for this wave's N columns) into registers
    v16h Bv[8];
    {
        const _Float16* wrow = Whh + (size_t)ncol * HID + khalf * 8;
#pragma unroll
        for (int c = 0; c < 8; ++c) {
            F t;
            t.q[0] = *(const uint4*)(wrow + c * 32);
            t.q[1] = *(const uint4*)(wrow + c * 32 + 16);
            Bv[c] = t.h16;
        }
    }

    // zero initial h, c
    for (int i = gtid; i < BATCH * HID; i += 64 * 256) {
        Hbuf[i] = (_Float16)0.0f;
        Cbuf[i] = 0.0f;
    }
    grid_sync(bcount, bphase, &lphase);

    for (int t = 0; t < TLEN; ++t) {
        // ---- gates tile: acc = P[t,:,:] + h @ Whh^T ----
        const _Float16* Pt = P + (size_t)t * BATCH * G4H;
        v8f acc;
#pragma unroll
        for (int v = 0; v < 8; ++v)
            acc[v] = (float)Pt[(size_t)(crow0 + v) * G4H + ncol];

        // preload all A fragments (h rows), then run 8 chained WMMAs back-to-back
        v16h Av[8];
        const _Float16* hrow = Hbuf + (size_t)mrow * HID + khalf * 8;
#pragma unroll
        for (int c = 0; c < 8; ++c) {
            F a;
            a.q[0] = *(const uint4*)(hrow + c * 32);
            a.q[1] = *(const uint4*)(hrow + c * 32 + 16);
            Av[c] = a.h16;
        }
#pragma unroll
        for (int c = 0; c < 8; ++c)
            acc = __builtin_amdgcn_wmma_f32_16x16x32_f16(
                false, Av[c], false, Bv[c], (short)0, acc, false, false);

#pragma unroll
        for (int v = 0; v < 8; ++v)
            Gates[(size_t)(crow0 + v) * G4H + ncol] = acc[v];

        grid_sync(bcount, bphase, &lphase);

        // ---- elementwise c/h update (gate order i, f, g, o) ----
        for (int i = gtid; i < BATCH * HID; i += 64 * 256) {
            int b = i >> 8, j = i & (HID - 1);
            const float* g = Gates + (size_t)b * G4H;
            float ig = sigm(g[j]);
            float fg = sigm(g[HID + j]);
            float gg = tanh_fast(g[2 * HID + j]);
            float og = sigm(g[3 * HID + j]);
            float cn = fg * Cbuf[i] + ig * gg;
            float hn = og * tanh_fast(cn);
            Cbuf[i] = cn;
            _Float16 hh = (_Float16)hn;
            Hbuf[i] = hh;
            Xout[((size_t)b * TLEN + t) * HID + j] = hh;
        }
        grid_sync(bcount, bphase, &lphase);
    }
}

// logits = h_T @ fc_w^T + fc_b  -> [128, 10] f32
__global__ void k_logits(const _Float16* __restrict__ H, const float* __restrict__ fcw,
                         const float* __restrict__ fcb, float* __restrict__ out) {
    for (int o = threadIdx.x; o < BATCH * NLAB; o += blockDim.x) {
        int b = o / NLAB, l = o % NLAB;
        float s = fcb[l];
        for (int k = 0; k < HID; ++k)
            s += (float)H[b * HID + k] * fcw[l * HID + k];
        out[o] = s;
    }
}

// ---------- launch ----------

extern "C" void kernel_launch(void* const* d_in, const int* in_sizes, int n_in,
                              void* d_out, int out_size, void* d_ws, size_t ws_size,
                              hipStream_t stream) {
    (void)in_sizes; (void)n_in; (void)out_size; (void)ws_size;
    const int*   ids  = (const int*)d_in[0];
    const float* emb  = (const float*)d_in[1];
    const float* wih  = (const float*)d_in[2];
    const float* whh  = (const float*)d_in[3];
    const float* bih  = (const float*)d_in[4];
    const float* bhh  = (const float*)d_in[5];
    const float* fcw  = (const float*)d_in[6];
    const float* fcb  = (const float*)d_in[7];
    float* out = (float*)d_out;

    char*  ws  = (char*)d_ws;
    size_t off = 0;
    auto carve = [&](size_t bytes) { char* p = ws + off; off = (off + bytes + 255) & ~(size_t)255; return p; };

    _Float16* Xa   = (_Float16*)carve((size_t)BT * HID * 2);
    _Float16* Xb   = (_Float16*)carve((size_t)BT * HID * 2);
    _Float16* P    = (_Float16*)carve((size_t)BT * G4H * 2);
    _Float16* WihH = (_Float16*)carve((size_t)LAYERS * G4H * HID * 2);
    _Float16* WhhH = (_Float16*)carve((size_t)LAYERS * G4H * HID * 2);
    float*    Bsum = (float*)carve((size_t)LAYERS * G4H * 4);
    _Float16* Hbuf = (_Float16*)carve((size_t)BATCH * HID * 2);
    float*    Cbuf = (float*)carve((size_t)BATCH * HID * 4);
    float*    Gts  = (float*)carve((size_t)BATCH * G4H * 4);
    unsigned* Bar  = (unsigned*)carve(256);

    // 1) embedding gather + f16 convert
    {
        size_t nthreads = (size_t)BT * EMB / 8;
        k_embed<<<dim3((unsigned)(nthreads / 256)), dim3(256), 0, stream>>>(ids, emb, Xa);
    }
    // 2) weight conversion + bias fusion
    {
        size_t n = (size_t)LAYERS * G4H * HID;
        k_cvtw<<<dim3((unsigned)((n + 255) / 256)), dim3(256), 0, stream>>>(
            wih, whh, bih, bhh, WihH, WhhH, Bsum);
    }
    // 3) layers
    _Float16* Xin  = Xa;
    _Float16* Xout = Xb;
    for (int l = 0; l < LAYERS; ++l) {
        k_xproj<<<dim3(BT / 16), dim3(256), 0, stream>>>(
            Xin, WihH + (size_t)l * G4H * HID, Bsum + (size_t)l * G4H, P);
        hipMemsetAsync(Bar, 0, 8, stream);
        k_scan<<<dim3(64), dim3(256), 0, stream>>>(
            P, WhhH + (size_t)l * G4H * HID, Hbuf, Cbuf, Gts, Xout, Bar);
        _Float16* t = Xin; Xin = Xout; Xout = t;
    }
    // 4) classifier head
    k_logits<<<dim3(1), dim3(256), 0, stream>>>(Hbuf, fcw, fcb, out);
}
